// QuantBottleneck_87016037417493
// MI455X (gfx1250) — matrix operand order repaired and checked by reading, not compiled, not run on hardware
//
#include <hip/hip_runtime.h>
#include <hip/hip_bf16.h>

typedef _Float16 v16h __attribute__((ext_vector_type(16)));
typedef _Float16 h8   __attribute__((ext_vector_type(8)));
typedef float    v8f  __attribute__((ext_vector_type(8)));
typedef int      v4i  __attribute__((ext_vector_type(4)));

#define AS1 __attribute__((address_space(1)))
#define AS3 __attribute__((address_space(3)))

#if defined(__HIP_DEVICE_COMPILE__) && __has_builtin(__builtin_amdgcn_global_load_async_to_lds_b128)
#define HAVE_ASYNC_LDS 1
#else
#define HAVE_ASYNC_LDS 0
#endif

// LDS row strides (halves): stride*2 bytes must be multiple of 16; stride/2 mod 64 == 4
#define ST1 136   // gemm tile: 32 rows x 128 K-halves (+8 pad)
#define ST2 264   // conv3x3 tile: 32 rows x 256 K-halves (+8 pad)

__device__ __forceinline__ v8f wmma32(v16h a, v16h b, v8f c) {
  return __builtin_amdgcn_wmma_f32_16x16x32_f16(false, a, false, b, (short)0, c,
                                                false, false);
}

// A-fragment per ISA layout: elems 0..7 = p[0..7], elems 8..15 = p[16..23]
__device__ __forceinline__ v16h load_frag_a(const _Float16* p) {
  h8 lo = *(const h8*)p;
  h8 hi = *(const h8*)(p + 16);
  return __builtin_shufflevector(lo, hi, 0,1,2,3,4,5,6,7,8,9,10,11,12,13,14,15);
}
// B-fragment per ISA layout: elems 0..7 = p[0..7], elems 8..15 = p[16..23]
__device__ __forceinline__ v16h load_frag_b(const _Float16* p) {
  h8 lo = *(const h8*)p;
  h8 hi = *(const h8*)(p + 16);
  return __builtin_shufflevector(lo, hi, 0,1,2,3,4,5,6,7,8,9,10,11,12,13,14,15);
}

// 16 bytes global -> LDS, async (ASYNCcnt) when available
__device__ __forceinline__ void copy16_g2l(const _Float16* gsrc, _Float16* ldst) {
#if HAVE_ASYNC_LDS
  __builtin_amdgcn_global_load_async_to_lds_b128(
      (AS1 v4i*)gsrc, (AS3 v4i*)ldst, 0, 0);
#else
  *(h8*)ldst = *(const h8*)gsrc;
#endif
}

__device__ __forceinline__ void wait_async_lds() {
#if HAVE_ASYNC_LDS
#if __has_builtin(__builtin_amdgcn_s_wait_asynccnt)
  __builtin_amdgcn_s_wait_asynccnt(0);
#else
  asm volatile("s_wait_asynccnt 0x0" ::: "memory");
#endif
#endif
}

// ---------------- weight blockwise quantization (int8 x per-block scale -> f16)
// out layout: [khw][cout][cin]  (cin contiguous = GEMM K)
__global__ void quant_w_kernel(const float* __restrict__ w, _Float16* __restrict__ o,
                               int Cout, int Cin, int KHW) {
  int t = blockIdx.x * blockDim.x + threadIdx.x;
  int nb = Cin >> 5;
  int total = Cout * nb * KHW;
  if (t >= total) return;
  int kk = t % KHW;
  int tmp = t / KHW;
  int cb = tmp % nb;
  int cout = tmp / nb;
  const float* base = w + ((size_t)(cout * Cin + cb * 32)) * KHW + kk;
  float mx = 0.f;
  for (int j = 0; j < 32; ++j) mx = fmaxf(mx, fabsf(base[(size_t)j * KHW]));
  float s = fmaxf(mx / 127.f, 1e-12f);
  float inv = 1.f / s;
  _Float16* ob = o + ((size_t)kk * Cout + cout) * Cin + cb * 32;
  for (int j = 0; j < 32; ++j) {
    float q = fminf(fmaxf(rintf(base[(size_t)j * KHW] * inv), -128.f), 127.f) * s;
    ob[j] = (_Float16)q;
  }
}

// ---------------- x: NCHW f32 -> NHWC f16, channel-fastest (coalesced stores)
__global__ void nchw2nhwc_kernel(const float* __restrict__ x, _Float16* __restrict__ xh) {
  long t = (long)blockIdx.x * blockDim.x + threadIdx.x;   // 64*196*1024 total
  if (t >= 64L * 196 * 1024) return;
  int c = (int)(t & 1023);
  long p = t >> 10;
  long n = p / 196;
  int hw = (int)(p - n * 196);
  xh[t] = (_Float16)x[(n * 1024 + c) * 196 + hw];
}

// ---------------- fused BN (+ residual) + ReLU + BFP-quant epilogue
template <bool FINAL>
__device__ __forceinline__ void epilogue32(
    v8f c00, v8f c01, v8f c10, v8f c11, long mbase, int nbase, int half, int l16,
    const float* __restrict__ g, const float* __restrict__ be,
    const float* __restrict__ mn, const float* __restrict__ vr,
    const float* __restrict__ resid, float* __restrict__ outF,
    _Float16* __restrict__ outH) {
  int col0 = nbase + l16, col1 = col0 + 16;
  float inv0 = g[col0] * rsqrtf(vr[col0] + 1e-5f);
  float bia0 = be[col0] - mn[col0] * inv0;
  float inv1 = g[col1] * rsqrtf(vr[col1] + 1e-5f);
  float bia1 = be[col1] - mn[col1] * inv1;
#pragma unroll
  for (int mi = 0; mi < 2; ++mi) {
    v8f d0 = mi ? c10 : c00;
    v8f d1 = mi ? c11 : c01;
#pragma unroll
    for (int r = 0; r < 8; ++r) {
      // C layout: VGPR r -> row r (lanes 0-15) / row r+8 (lanes 16-31)
      long row = mbase + mi * 16 + (half ? 8 : 0) + r;
      float v0 = d0[r] * inv0 + bia0;
      float v1 = d1[r] * inv1 + bia1;
      long nimg = row / 196;
      int hw = (int)(row - nimg * 196);
      if (FINAL) {
        v0 += resid[(nimg * 1024 + col0) * 196 + hw];
        v1 += resid[(nimg * 1024 + col1) * 196 + hw];
      }
      v0 = fmaxf(v0, 0.f);
      v1 = fmaxf(v1, 0.f);
      float mx = fmaxf(v0, v1);            // post-ReLU => abs == value
#pragma unroll
      for (int off = 1; off < 16; off <<= 1)
        mx = fmaxf(mx, __shfl_xor(mx, off, 32));
      float e = floorf(log2f(fmaxf(mx, 1e-12f)));
      float s = exp2f(e - 6.f);            // 2^(e+2-bits), bits=8
      float rs = exp2f(6.f - e);
      float q0 = fminf(fmaxf(rintf(v0 * rs), -128.f), 127.f) * s;
      float q1 = fminf(fmaxf(rintf(v1 * rs), -128.f), 127.f) * s;
      if (FINAL) {
        outF[(nimg * 1024 + col0) * 196 + hw] = q0;
        outF[(nimg * 1024 + col1) * 196 + hw] = q1;
      } else {
        outH[row * 256 + col0] = (_Float16)q0;
        outH[row * 256 + col1] = (_Float16)q1;
      }
    }
  }
}

// ---------------- 1x1 conv as GEMM (M=12544 pixels), K-tile=128 staged in LDS
template <int K, bool FINAL>
__global__ __launch_bounds__(256) void gemm_bfp_kernel(
    const _Float16* __restrict__ A, const _Float16* __restrict__ W,
    const float* __restrict__ g, const float* __restrict__ be,
    const float* __restrict__ mn, const float* __restrict__ vr,
    const float* __restrict__ resid, float* __restrict__ outF,
    _Float16* __restrict__ outH) {
  __shared__ __align__(16) _Float16 As[32 * ST1];
  const int tid = threadIdx.x;
  const int wave = tid >> 5, lane = tid & 31;
  const int half = lane >> 4, l16 = lane & 15;
  const int nbase = blockIdx.y * 256 + wave * 32;
  const long mbase = (long)blockIdx.x * 32;

  // fill geometry: thread owns (row = tid/8, 16 halves at seg*16)
  const int frow = tid >> 3, fseg = tid & 7;
  const _Float16* arow = A + (mbase + frow) * K + fseg * 16;
  _Float16* fdst = &As[frow * ST1 + fseg * 16];

  v8f c00 = {}, c01 = {}, c10 = {}, c11 = {};
  const int kb_a = half ? 8 : 0;
  const int kb_b = half ? 8 : 0;
  const _Float16* w0p = W + (size_t)(nbase + l16) * K + kb_b;
  const _Float16* w1p = w0p + (size_t)16 * K;

  for (int k0 = 0; k0 < K; k0 += 128) {
    __syncthreads();
    copy16_g2l(arow + k0, fdst);
    copy16_g2l(arow + k0 + 8, fdst + 8);
    wait_async_lds();
    __syncthreads();
    if (k0 + 128 < K) {
      __builtin_prefetch(w0p + k0 + 128, 0, 3);
      __builtin_prefetch(w1p + k0 + 128, 0, 3);
    }
#pragma unroll
    for (int kc = 0; kc < 128; kc += 32) {
      v16h a0 = load_frag_a(&As[l16 * ST1 + kc + kb_a]);
      v16h a1 = load_frag_a(&As[(16 + l16) * ST1 + kc + kb_a]);
      v16h b0 = load_frag_b(w0p + k0 + kc);
      v16h b1 = load_frag_b(w1p + k0 + kc);
      c00 = wmma32(a0, b0, c00);
      c01 = wmma32(a0, b1, c01);
      c10 = wmma32(a1, b0, c10);
      c11 = wmma32(a1, b1, c11);
    }
  }
  epilogue32<FINAL>(c00, c01, c10, c11, mbase, nbase, half, l16, g, be, mn, vr,
                    resid, outF, outH);
}

// ---------------- 3x3 conv (pad 1): stage full K=256 tile per tap in LDS
__global__ __launch_bounds__(256) void conv3x3_bfp_kernel(
    const _Float16* __restrict__ A /* NHWC 256 */,
    const _Float16* __restrict__ W /* [9][256][256] */,
    const float* __restrict__ g, const float* __restrict__ be,
    const float* __restrict__ mn, const float* __restrict__ vr,
    _Float16* __restrict__ outH) {
  __shared__ __align__(16) _Float16 As[32 * ST2];   // ~16.9 KB of 320 KB
  const int tid = threadIdx.x;
  const int wave = tid >> 5, lane = tid & 31;
  const int half = lane >> 4, l16 = lane & 15;
  const int nbase = wave * 32;
  const long mbase = (long)blockIdx.x * 32;

  // fill geometry: thread owns (row = tid/8, 32 halves at seg*32); hoisted decode
  const int frow = tid >> 3, fseg = tid & 7;
  const long fp = mbase + frow;
  const long fn = fp / 196;
  const int fhw = (int)(fp - fn * 196);
  const int fh = fhw / 14, fw = fhw - fh * 14;
  const _Float16* arow0 = A + (fn * 196) * 256 + fseg * 32;
  _Float16* fdst = &As[frow * ST2 + fseg * 32];

  v8f c00 = {}, c01 = {}, c10 = {}, c11 = {};
  const int kb_a = half ? 8 : 0;
  const int kb_b = half ? 8 : 0;

  for (int tap = 0; tap < 9; ++tap) {
    const int dh = tap / 3 - 1, dw = tap % 3 - 1;
    const int h = fh + dh, w = fw + dw;
    const bool valid = ((unsigned)h < 14u) && ((unsigned)w < 14u);
    const _Float16* src = arow0 + (h * 14 + w) * 256;

    __syncthreads();
    if (valid) {
#pragma unroll
      for (int i = 0; i < 4; ++i) copy16_g2l(src + i * 8, fdst + i * 8);
    } else {
      h8 z = {};
#pragma unroll
      for (int i = 0; i < 4; ++i) *(h8*)(fdst + i * 8) = z;
    }
    wait_async_lds();
    __syncthreads();

    const _Float16* w0p = W + (size_t)tap * 256 * 256 + (size_t)(nbase + l16) * 256 + kb_b;
    const _Float16* w1p = w0p + (size_t)16 * 256;
    __builtin_prefetch(w0p, 0, 3);
#pragma unroll
    for (int kc = 0; kc < 256; kc += 32) {
      v16h a0 = load_frag_a(&As[l16 * ST2 + kc + kb_a]);
      v16h a1 = load_frag_a(&As[(16 + l16) * ST2 + kc + kb_a]);
      v16h b0 = load_frag_b(w0p + kc);
      v16h b1 = load_frag_b(w1p + kc);
      c00 = wmma32(a0, b0, c00);
      c01 = wmma32(a0, b1, c01);
      c10 = wmma32(a1, b0, c10);
      c11 = wmma32(a1, b1, c11);
    }
  }
  epilogue32<false>(c00, c01, c10, c11, mbase, nbase, half, l16, g, be, mn, vr,
                    nullptr, nullptr, outH);
}

extern "C" void kernel_launch(void* const* d_in, const int* in_sizes, int n_in,
                              void* d_out, int out_size, void* d_ws, size_t ws_size,
                              hipStream_t stream) {
  (void)in_sizes; (void)n_in; (void)out_size; (void)ws_size;
  const float* x  = (const float*)d_in[0];
  const float* w1 = (const float*)d_in[1];
  const float* w2 = (const float*)d_in[2];
  const float* w3 = (const float*)d_in[3];
  const float* g1 = (const float*)d_in[4];
  const float* b1 = (const float*)d_in[5];
  const float* m1 = (const float*)d_in[6];
  const float* v1 = (const float*)d_in[7];
  const float* g2 = (const float*)d_in[8];
  const float* b2 = (const float*)d_in[9];
  const float* m2 = (const float*)d_in[10];
  const float* v2 = (const float*)d_in[11];
  const float* g3 = (const float*)d_in[12];
  const float* b3 = (const float*)d_in[13];
  const float* m3 = (const float*)d_in[14];
  const float* v3 = (const float*)d_in[15];
  float* out = (float*)d_out;

  char* ws = (char*)d_ws;
  size_t o = 0;
  auto take = [&](size_t bytes) {
    char* p = ws + o;
    o = (o + bytes + 255) & ~(size_t)255;
    return p;
  };
  _Float16* xh   = (_Float16*)take(64L * 196 * 1024 * 2);  // NHWC f16 x
  _Float16* act1 = (_Float16*)take(64L * 196 * 256 * 2);
  _Float16* act2 = (_Float16*)take(64L * 196 * 256 * 2);
  _Float16* W1h  = (_Float16*)take(256L * 1024 * 2);
  _Float16* W2h  = (_Float16*)take(9L * 256 * 256 * 2);
  _Float16* W3h  = (_Float16*)take(1024L * 256 * 2);

  // 1) blockwise weight quantization -> dequantized f16, repacked
  quant_w_kernel<<<32, 256, 0, stream>>>(w1, W1h, 256, 1024, 1);
  quant_w_kernel<<<72, 256, 0, stream>>>(w2, W2h, 256, 256, 9);
  quant_w_kernel<<<32, 256, 0, stream>>>(w3, W3h, 1024, 256, 1);

  // 2) x -> NHWC f16
  nchw2nhwc_kernel<<<50176, 256, 0, stream>>>(x, xh);

  // 3) conv1 1x1 (K=1024, N=256) + BN1/ReLU/BFP
  gemm_bfp_kernel<1024, false><<<dim3(392, 1), 256, 0, stream>>>(
      xh, W1h, g1, b1, m1, v1, nullptr, nullptr, act1);

  // 4) conv2 3x3 pad1 + BN2/ReLU/BFP
  conv3x3_bfp_kernel<<<392, 256, 0, stream>>>(act1, W2h, g2, b2, m2, v2, act2);

  // 5) conv3 1x1 (K=256, N=1024) + BN3 + residual + ReLU + BFP -> f32 NCHW
  gemm_bfp_kernel<256, true><<<dim3(392, 4), 256, 0, stream>>>(
      act2, W3h, g3, b3, m3, v3, x, out, nullptr);
}
